// FeatureGeneratorSRU_79869211836596
// MI455X (gfx1250) — compile-verified, matrix operand order checked by reading, not run on hardware
//
#include <hip/hip_runtime.h>
#include <hip/hip_bf16.h>
#include <math.h>

// ---------------------------------------------------------------------------
// FeatureGeneratorSRU for MI455X (gfx1250, wave32, WMMA).
// Pipeline per layer: bf16 WMMA GEMM (U = X @ W) -> f32 sequential scan.
// Final: time-mean + layernorm.
// ---------------------------------------------------------------------------

typedef __bf16   bf16_t;
typedef __bf16   v16bf __attribute__((ext_vector_type(16)));
typedef float    v8f   __attribute__((ext_vector_type(8)));

#define SRU_L 1024
#define SRU_B 64
#define SRU_H 128
#define SRU_M (SRU_L * SRU_B)   // 65536 rows in every GEMM

// ---------------------------------------------------------------------------
// f32 -> bf16 elementwise convert
// ---------------------------------------------------------------------------
__global__ __launch_bounds__(256) void cvt_f32_bf16(const float* __restrict__ src,
                                                    bf16_t* __restrict__ dst, int n) {
    int i = blockIdx.x * 256 + threadIdx.x;
    if (i < n) dst[i] = (bf16_t)src[i];
}

// ---------------------------------------------------------------------------
// W (K,N) f32 row-major  ->  Wt (N,K) bf16 row-major (for B-fragment loads)
// ---------------------------------------------------------------------------
__global__ __launch_bounds__(256) void transpose_w_bf16(const float* __restrict__ W,
                                                        bf16_t* __restrict__ Wt,
                                                        int K, int N) {
    int i = blockIdx.x * 256 + threadIdx.x;
    if (i < N * K) {
        int n = i / K;
        int k = i - n * K;
        Wt[i] = (bf16_t)W[(size_t)k * N + n];
    }
}

// ---------------------------------------------------------------------------
// WMMA bf16 GEMM:  C(M,N) f32 = A(M,K) bf16  x  Bt(N,K) bf16 (B transposed).
// Block = 256 threads = 8 waves (2x4); block tile 128M x 256N.
// Wave tile 64M x 64N as a 4x4 grid of 16x16 v_wmma_f32_16x16x32_bf16 accums:
// 16 WMMAs per 8 fragment loads per K-step.
//
// A fragment (16x32, 8 VGPRs): lane l<16 holds row (l), K[kb..kb+16);
//                              lane l>=16 holds row (l-16), K[kb+16..kb+32).
// B fragment (32x16): identical per-lane pattern on Wt rows (columns of W).
// C/D 16x16 f32: VGPR r, lane l -> (M = r + 8*(l>=16), N = l&15).
//
// NOTE: fragment addresses are full GEP expressions on the __restrict__
// kernel args (not loop-carried pointer variables) so address-space
// inference resolves them to GLOBAL loads (global_load_b128), not FLAT
// loads that would tie up the LDS path / DScnt. LSR still strength-reduces
// the kb recurrence to pointer increments.
// ---------------------------------------------------------------------------
__global__ __launch_bounds__(256) void wmma_gemm_bf16(const bf16_t* __restrict__ A,
                                                      const bf16_t* __restrict__ Bt,
                                                      float* __restrict__ C,
                                                      int N, int K) {
    const int lane = threadIdx.x & 31;
    const int wave = threadIdx.x >> 5;     // 0..7
    const int wm   = wave & 1;             // 2 M slots of 64 rows
    const int wn   = wave >> 1;            // 4 N slots of 64 cols
    const int m0   = blockIdx.y * 128 + wm * 64;
    const int n0   = blockIdx.x * 256 + wn * 64;
    const int lr   = lane & 15;            // row within fragment
    const int lk   = (lane >> 4) << 4;     // 0 or 16: K-half select

    v8f acc[4][4];
    const v8f vzero = {0.f, 0.f, 0.f, 0.f, 0.f, 0.f, 0.f, 0.f};
#pragma unroll
    for (int i = 0; i < 4; ++i)
#pragma unroll
        for (int j = 0; j < 4; ++j) acc[i][j] = vzero;

    for (int kb = 0; kb < K; kb += 32) {
        v16bf afrag[4], bfrag[4];
#pragma unroll
        for (int i = 0; i < 4; ++i)
            afrag[i] = *(const v16bf*)(A + (size_t)(m0 + i * 16 + lr) * K + kb + lk);
#pragma unroll
        for (int j = 0; j < 4; ++j)
            bfrag[j] = *(const v16bf*)(Bt + (size_t)(n0 + j * 16 + lr) * K + kb + lk);
#pragma unroll
        for (int i = 0; i < 4; ++i)
#pragma unroll
            for (int j = 0; j < 4; ++j)
                acc[i][j] = __builtin_amdgcn_wmma_f32_16x16x32_bf16(
                    false, afrag[i], false, bfrag[j], (short)0, acc[i][j], false, false);
    }

    const int mhalf = (lane >> 4) << 3;   // +8 rows for upper lane half
#pragma unroll
    for (int i = 0; i < 4; ++i) {
        const int mb = m0 + i * 16 + mhalf;
#pragma unroll
        for (int j = 0; j < 4; ++j) {
            const int n = n0 + j * 16 + lr;
#pragma unroll
            for (int r = 0; r < 8; ++r)
                C[(size_t)(mb + r) * N + n] = acc[i][j][r];
        }
    }
}

// ---------------------------------------------------------------------------
// SRU bidirectional scan. One thread per (b, dir, h) state, loops over L.
// U layout: (L, B, Nld) with column = dir*H*k + h*k + kk.
// Backward direction processes t = L-1-s and writes at the same t, which
// implements both the time reversal and the un-reversal of the reference.
// Writes layer output as f32 (L,B,2H) and bf16 (for the next GEMM).
// ---------------------------------------------------------------------------
__global__ __launch_bounds__(256) void sru_scan(const float* __restrict__ U,
                                                const float* __restrict__ xin,
                                                const float* __restrict__ bias,
                                                int k, int Nld,
                                                float* __restrict__ xout,
                                                bf16_t* __restrict__ xout_bf) {
    const int idx = blockIdx.x * 256 + threadIdx.x;   // 0..16383
    const int b   = idx >> 8;
    const int rem = idx & 255;
    const int dir = rem >> 7;
    const int h   = rem & 127;
    const int col = dir * (SRU_H * k) + h * k;
    const int oc  = dir * SRU_H + h;

    const float bfv = bias[dir * SRU_H + h];
    const float brv = bias[2 * SRU_H + dir * SRU_H + h];

    const float selu_s  = 1.0507009873554805f;           // scale
    const float selu_as = 1.7580993408473766f;           // scale*alpha

    float c = 0.0f;
    for (int s = 0; s < SRU_L; ++s) {
        const int t = dir ? (SRU_L - 1 - s) : s;
        const size_t row = (size_t)(t * SRU_B + b);
        const float* u = U + row * Nld + col;
        float x0 = u[0];
        float f  = 1.0f / (1.0f + expf(-(u[1] + bfv)));
        float r  = 1.0f / (1.0f + expf(-(u[2] + brv)));
        float xp = (k == 4) ? u[3] : xin[row * (2 * SRU_H) + oc];
        c = f * c + (1.0f - f) * x0;
        float sel = (c > 0.0f) ? selu_s * c : selu_as * (expf(c) - 1.0f);
        float hv  = r * sel + (1.0f - r) * xp;
        size_t oi = row * (2 * SRU_H) + oc;
        xout[oi]    = hv;
        xout_bf[oi] = (bf16_t)hv;
    }
}

// ---------------------------------------------------------------------------
// feats = mean over L of x4 (L,B,256); then layernorm over the 256 features.
// One block per batch row, 256 threads (one per feature).
// ---------------------------------------------------------------------------
__global__ __launch_bounds__(256) void mean_layernorm(const float* __restrict__ x4,
                                                      const float* __restrict__ gamma,
                                                      const float* __restrict__ beta,
                                                      float* __restrict__ out) {
    const int b = blockIdx.x;
    const int j = threadIdx.x;
    float s = 0.0f;
    for (int t = 0; t < SRU_L; ++t)
        s += x4[(size_t)(t * SRU_B + b) * 256 + j];
    const float feat = s * (1.0f / (float)SRU_L);

    __shared__ float red[256];
    red[j] = feat;
    __syncthreads();
#pragma unroll
    for (int st = 128; st > 0; st >>= 1) {
        if (j < st) red[j] += red[j + st];
        __syncthreads();
    }
    const float mu = red[0] * (1.0f / 256.0f);
    __syncthreads();
    const float d = feat - mu;
    red[j] = d * d;
    __syncthreads();
#pragma unroll
    for (int st = 128; st > 0; st >>= 1) {
        if (j < st) red[j] += red[j + st];
        __syncthreads();
    }
    const float var = red[0] * (1.0f / 256.0f);
    out[b * 256 + j] = d * rsqrtf(var + 1e-5f) * gamma[j] + beta[j];
}

// ---------------------------------------------------------------------------
// Host launcher
// Input dict order: input, W0, b0, gamma, beta, W1, b1, W2, b2, W3, b3
// ---------------------------------------------------------------------------
extern "C" void kernel_launch(void* const* d_in, const int* in_sizes, int n_in,
                              void* d_out, int out_size, void* d_ws, size_t ws_size,
                              hipStream_t stream) {
    const float* x0    = (const float*)d_in[0];
    const float* W0    = (const float*)d_in[1];
    const float* b0    = (const float*)d_in[2];
    const float* gamma = (const float*)d_in[3];
    const float* beta  = (const float*)d_in[4];
    const float* W1    = (const float*)d_in[5];
    const float* b1    = (const float*)d_in[6];
    const float* W2    = (const float*)d_in[7];
    const float* b2    = (const float*)d_in[8];
    const float* W3    = (const float*)d_in[9];
    const float* b3    = (const float*)d_in[10];
    float* out = (float*)d_out;

    char*  ws  = (char*)d_ws;
    size_t off = 0;
    auto carve = [&](size_t bytes) -> char* {
        char* p = ws + off;
        off += (bytes + 255) & ~(size_t)255;
        return p;
    };

    const size_t M = SRU_M;
    float*  U   = (float*)carve(M * 1024 * sizeof(float));      // 256 MB, reused per layer
    float*  xA  = (float*)carve(M * 256 * sizeof(float));       // 64 MB
    float*  xB  = (float*)carve(M * 256 * sizeof(float));       // 64 MB
    bf16_t* xb  = (bf16_t*)carve(M * 256 * sizeof(bf16_t));     // 32 MB
    bf16_t* wt0 = (bf16_t*)carve((size_t)1024 * 128 * sizeof(bf16_t));
    bf16_t* wt1 = (bf16_t*)carve((size_t)768 * 256 * sizeof(bf16_t));
    bf16_t* wt2 = (bf16_t*)carve((size_t)768 * 256 * sizeof(bf16_t));
    bf16_t* wt3 = (bf16_t*)carve((size_t)768 * 256 * sizeof(bf16_t));
    (void)ws_size; (void)in_sizes; (void)n_in; (void)out_size;

    // --- weight transposes + input conversion (tiny) ---
    transpose_w_bf16<<<(1024 * 128 + 255) / 256, 256, 0, stream>>>(W0, wt0, 128, 1024);
    transpose_w_bf16<<<(768 * 256 + 255) / 256, 256, 0, stream>>>(W1, wt1, 256, 768);
    transpose_w_bf16<<<(768 * 256 + 255) / 256, 256, 0, stream>>>(W2, wt2, 256, 768);
    transpose_w_bf16<<<(768 * 256 + 255) / 256, 256, 0, stream>>>(W3, wt3, 256, 768);
    cvt_f32_bf16<<<(int)((M * 128 + 255) / 256), 256, 0, stream>>>(x0, xb, (int)(M * 128));

    const dim3 blk(256);
    const int scan_blocks = (SRU_B * 2 * SRU_H) / 256;   // 64

    // --- layer 0: k=4, N=1024, K=128 ---
    wmma_gemm_bf16<<<dim3(1024 / 256, (unsigned)(M / 128)), blk, 0, stream>>>(xb, wt0, U, 1024, 128);
    sru_scan<<<scan_blocks, 256, 0, stream>>>(U, nullptr, b0, 4, 1024, xA, xb);

    // --- layer 1: k=3, N=768, K=256 ---
    wmma_gemm_bf16<<<dim3(768 / 256, (unsigned)(M / 128)), blk, 0, stream>>>(xb, wt1, U, 768, 256);
    sru_scan<<<scan_blocks, 256, 0, stream>>>(U, xA, b1, 3, 768, xB, xb);

    // --- layer 2 ---
    wmma_gemm_bf16<<<dim3(768 / 256, (unsigned)(M / 128)), blk, 0, stream>>>(xb, wt2, U, 768, 256);
    sru_scan<<<scan_blocks, 256, 0, stream>>>(U, xB, b2, 3, 768, xA, xb);

    // --- layer 3 ---
    wmma_gemm_bf16<<<dim3(768 / 256, (unsigned)(M / 128)), blk, 0, stream>>>(xb, wt3, U, 768, 256);
    sru_scan<<<scan_blocks, 256, 0, stream>>>(U, xA, b3, 3, 768, xB, xb);

    // --- mean over time + layernorm ---
    mean_layernorm<<<SRU_B, 256, 0, stream>>>(xB, gamma, beta, out);
}
